// BahdanauAttn_17506286698677
// MI455X (gfx1250) — matrix operand order repaired and checked by reading, not compiled, not run on hardware
//
#include <hip/hip_runtime.h>
#include <hip/hip_bf16.h>

#define ENC  1024
#define DEC  1024
#define ADIM 512
#define BB   64
#define TT   2048

typedef __attribute__((ext_vector_type(16))) __bf16 v16bf;
typedef __attribute__((ext_vector_type(8)))  float  v8f;

static __device__ __forceinline__ unsigned short bfbits(float f) {
  unsigned u = __builtin_bit_cast(unsigned, f);
  return (unsigned short)((u + 0x7fffu + ((u >> 16) & 1u)) >> 16);  // RNE
}

// ---------------- Kernel 0: Wh f32 [512,1024] -> bf16 (row-major) ----------------
__global__ __launch_bounds__(256)
void cvt_wh_kernel(const float* __restrict__ Wh, unsigned short* __restrict__ whb) {
  const int i = blockIdx.x * 256 + threadIdx.x;     // float4 index
  const float4 x = ((const float4*)Wh)[i];
  unsigned p0 = (unsigned)bfbits(x.x) | ((unsigned)bfbits(x.y) << 16);
  unsigned p1 = (unsigned)bfbits(x.z) | ((unsigned)bfbits(x.w) << 16);
  *(uint2*)(whb + (size_t)i * 4) = make_uint2(p0, p1);
}

// ---------------- Kernel 1: q[b,a] = s[b,:] . Ws[a,:] + bs[a] ----------------
__global__ __launch_bounds__(256)
void qproj_kernel(const float* __restrict__ s, const float* __restrict__ Ws,
                  const float* __restrict__ bs, float* __restrict__ q) {
  int idx = blockIdx.x * 256 + threadIdx.x;         // 0 .. B*ADIM-1
  int b = idx >> 9;
  int a = idx & (ADIM - 1);
  const float4* sv = (const float4*)(s + (size_t)b * DEC);
  const float4* wv = (const float4*)(Ws + (size_t)a * DEC);
  float acc = 0.f;
#pragma unroll 4
  for (int i = 0; i < DEC / 4; ++i) {
    float4 x = sv[i], y = wv[i];
    acc += x.x * y.x + x.y * y.y + x.z * y.z + x.w * y.w;
  }
  q[idx] = acc + bs[a];
}

// ---------------- Kernel 2: fused GEMM + tanh + v-dot -> e[b,t] ----------------
// Block: 256 threads = 8 waves; covers 64 rows of T x columns 0-255 (16 col-tiles).
// Wave w: mt-pair (w>>2)*2 (2 row-tiles, B reused across both) and
// column quarter w&3 (4 col-tiles). A staged to LDS as bf16 in 4 K-phases.
#define KPH 256                         // K per phase
__global__ __launch_bounds__(256)
void scores_kernel(const float* __restrict__ H, const unsigned short* __restrict__ whb,
                   const float* __restrict__ q, const float* __restrict__ v,
                   const unsigned char* __restrict__ mask, float* __restrict__ e) {
  __shared__ unsigned short hsh[64 * KPH];   // 32 KB: A phase tile, row-major
  __shared__ float lds_e[64];

  const int b   = blockIdx.y;
  const int t0  = blockIdx.x * 64;
  const int tid = threadIdx.x;
  if (tid < 64) lds_e[tid] = 0.f;

  const int w      = tid >> 5;
  const int lane   = tid & 31;
  const int half   = lane >> 4;       // 0: lanes 0-15, 1: lanes 16-31
  const int l16    = lane & 15;
  const int mtbase = (w >> 2) * 2;    // row tiles mtbase, mtbase+1
  const int nq     = w & 3;           // column quarter -> col tiles [4nq,4nq+4)

  // ISA layouts: 16-bit A 16x32 -> lanes<16 hold K 0-7 & 16-23, lanes>=16 +8;
  // B 32x16 -> same as A: lanes<16 hold K 0-7 & 16-23, lanes>=16 +8.
  const int abase = half ? 8 : 0;
  const int bbase = half ? 8 : 0;

  const unsigned short* bptr[4];
#pragma unroll
  for (int j = 0; j < 4; ++j)
    bptr[j] = whb + (size_t)((nq * 4 + j) * 16 + l16) * ENC + bbase;

  v8f acc[4][2];
#pragma unroll
  for (int j = 0; j < 4; ++j)
#pragma unroll
    for (int m = 0; m < 2; ++m) acc[j][m] = v8f{};

  const float* hbase = H + ((size_t)b * TT + t0) * ENC;

#pragma unroll 1
  for (int p = 0; p < ENC / KPH; ++p) {
    const int k0p = p * KPH;

    // ---- stage phase: 64 rows x 256 K, f32 -> bf16 into LDS ----
    for (int idx = tid; idx < 64 * (KPH / 4); idx += 256) {   // 4096 float4s
      const int r  = idx >> 6;          // row 0..63
      const int c4 = idx & 63;          // float4 within phase row
      float4 x = *(const float4*)(hbase + (size_t)r * ENC + k0p + c4 * 4);
      unsigned p0 = (unsigned)bfbits(x.x) | ((unsigned)bfbits(x.y) << 16);
      unsigned p1 = (unsigned)bfbits(x.z) | ((unsigned)bfbits(x.w) << 16);
      *(uint2*)(hsh + r * KPH + c4 * 4) = make_uint2(p0, p1);
    }
    __syncthreads();

    // ---- compute: 8 k-steps of 32; 8 WMMAs per step (4 nt x 2 mt) ----
#pragma unroll 1
    for (int kk = 0; kk < KPH; kk += 32) {
      union { uint4 u[2]; v16bf v; } af[2];
#pragma unroll
      for (int m = 0; m < 2; ++m) {
        const unsigned short* arow =
            hsh + ((mtbase + m) * 16 + l16) * KPH + kk + abase;
        af[m].u[0] = *(const uint4*)(arow);        // ds_load_b128
        af[m].u[1] = *(const uint4*)(arow + 16);   // ds_load_b128
      }
#pragma unroll
      for (int j = 0; j < 4; ++j) {
        union { uint4 u[2]; v16bf v; } bf;
        bf.u[0] = *(const uint4*)(bptr[j] + k0p + kk);      // global_load_b128
        bf.u[1] = *(const uint4*)(bptr[j] + k0p + kk + 16); // global_load_b128
#pragma unroll
        for (int m = 0; m < 2; ++m)
          acc[j][m] = __builtin_amdgcn_wmma_f32_16x16x32_bf16(
              false, af[m].v, false, bf.v, (short)0, acc[j][m], false, false);
      }
    }
    __syncthreads();   // protect LDS before next phase overwrite
  }

  // ---- epilogue: + q[b,col], tanh, * v[col], row-sum across columns ----
  float partial[2][8];
#pragma unroll
  for (int m = 0; m < 2; ++m)
#pragma unroll
    for (int i = 0; i < 8; ++i) partial[m][i] = 0.f;

#pragma unroll
  for (int j = 0; j < 4; ++j) {
    const int col = (nq * 4 + j) * 16 + l16;
    const float qv = q[b * ADIM + col];
    const float vv = v[col];
#pragma unroll
    for (int m = 0; m < 2; ++m)
#pragma unroll
      for (int i = 0; i < 8; ++i)
        partial[m][i] += tanhf(acc[j][m][i] + qv) * vv;
  }

  // C/D layout: VGPR i, lanes 0-15 -> row i, lanes 16-31 -> row 8+i (cols = lane%16).
#pragma unroll
  for (int m = 0; m < 2; ++m)
#pragma unroll
    for (int i = 0; i < 8; ++i) {
      float pp = partial[m][i];
      pp += __shfl_xor(pp, 1, 32);
      pp += __shfl_xor(pp, 2, 32);
      pp += __shfl_xor(pp, 4, 32);
      pp += __shfl_xor(pp, 8, 32);
      if (l16 == 0)
        atomicAdd(&lds_e[(mtbase + m) * 16 + half * 8 + i], pp);  // ds_add_f32
    }
  __syncthreads();

  if (tid < 64) {
    const int t = t0 + tid;
    const unsigned char mk = mask[(size_t)b * TT + t];
    e[(size_t)b * TT + t] = mk ? lds_e[tid] : -3.402823466e38f;
  }
}

// ---------------- Kernel 3: softmax over T per batch ----------------
__global__ __launch_bounds__(256)
void softmax_kernel(const float* __restrict__ e, float* __restrict__ a) {
  __shared__ float red[256];
  const int b = blockIdx.x;
  const float* er = e + (size_t)b * TT;
  float m = -3.402823466e38f;
  for (int t = threadIdx.x; t < TT; t += 256) m = fmaxf(m, er[t]);
  red[threadIdx.x] = m;
  __syncthreads();
  for (int sft = 128; sft > 0; sft >>= 1) {
    if (threadIdx.x < sft)
      red[threadIdx.x] = fmaxf(red[threadIdx.x], red[threadIdx.x + sft]);
    __syncthreads();
  }
  m = red[0];
  __syncthreads();
  float sum = 0.f;
  for (int t = threadIdx.x; t < TT; t += 256) sum += __expf(er[t] - m);
  red[threadIdx.x] = sum;
  __syncthreads();
  for (int sft = 128; sft > 0; sft >>= 1) {
    if (threadIdx.x < sft) red[threadIdx.x] += red[threadIdx.x + sft];
    __syncthreads();
  }
  const float inv = 1.f / red[0];
  float* ar = a + (size_t)b * TT;
  for (int t = threadIdx.x; t < TT; t += 256) ar[t] = __expf(er[t] - m) * inv;
}

// ---------------- Kernel 4: ctx[b,e] = sum_t a[b,t] * H[b,t,e] ----------------
__global__ __launch_bounds__(256)
void ctx_kernel(const float* __restrict__ H, const float* __restrict__ a,
                float* __restrict__ ctx) {
  const int b = blockIdx.y;
  const int ecol = blockIdx.x * 256 + threadIdx.x;   // 0..1023
  const float* hb = H + (size_t)b * TT * ENC + ecol;
  const float* ab = a + (size_t)b * TT;
  float acc = 0.f;
#pragma unroll 4
  for (int t = 0; t < TT; ++t) acc += ab[t] * hb[(size_t)t * ENC];
  ctx[(size_t)b * ENC + ecol] = acc;
}

extern "C" void kernel_launch(void* const* d_in, const int* in_sizes, int n_in,
                              void* d_out, int out_size, void* d_ws, size_t ws_size,
                              hipStream_t stream) {
  (void)in_sizes; (void)n_in; (void)out_size; (void)ws_size;
  const float* H  = (const float*)d_in[0];                   // [64,2048,1024]
  const float* s  = (const float*)d_in[1];                   // [64,1024]
  const unsigned char* mask = (const unsigned char*)d_in[2]; // [64,2048] bool
  const float* Wh = (const float*)d_in[3];                   // [512,1024]
  const float* Ws = (const float*)d_in[4];                   // [512,1024]
  const float* bs = (const float*)d_in[5];                   // [512]
  const float* v  = (const float*)d_in[6];                   // [512]

  float* ctx = (float*)d_out;                    // output 0: [64,1024]
  float* a   = (float*)d_out + BB * ENC;         // output 1: [64,2048]

  // workspace layout: q (128KB) | e (512KB) | Wh_bf16 (1MB)
  float* q = (float*)d_ws;
  float* e = (float*)((char*)d_ws + (size_t)BB * ADIM * sizeof(float));
  unsigned short* whb =
      (unsigned short*)((char*)d_ws + (size_t)BB * ADIM * sizeof(float)
                                    + (size_t)BB * TT * sizeof(float));

  cvt_wh_kernel<<<(ADIM * ENC / 4) / 256, 256, 0, stream>>>(Wh, whb);
  qproj_kernel<<<(BB * ADIM) / 256, 256, 0, stream>>>(s, Ws, bs, q);
  scores_kernel<<<dim3(TT / 64, BB), 256, 0, stream>>>(H, whb, q, v, mask, e);
  softmax_kernel<<<BB, 256, 0, stream>>>(e, a);
  ctx_kernel<<<dim3(ENC / 256, BB), 256, 0, stream>>>(H, a, ctx);
}
